// RSLDS_55001351192834
// MI455X (gfx1250) — matrix operand order repaired and checked by reading, not compiled, not run on hardware
//
#include <hip/hip_runtime.h>

typedef __attribute__((ext_vector_type(16))) __bf16 v16bf;
typedef __attribute__((ext_vector_type(8)))  __bf16 v8bf;
typedef __attribute__((ext_vector_type(8)))  float  v8f;

#define TT  1000
#define BB  16
#define TB  16000   /* B*T */
#define NN  32
#define KK  8
#define HH  8
#define RR  128
#define G3  384
#define IN_ 256

__device__ __forceinline__ float sigmoidf_(float x) { return 1.f / (1.f + __expf(-x)); }
__device__ __forceinline__ float4 ld4(const float* p) { return *(const float4*)p; }
__device__ __forceinline__ void st4(float* p, float4 v) { *(float4*)p = v; }
__device__ __forceinline__ void put4(v16bf& v, int i, float4 x) {
  v[i] = (__bf16)x.x; v[i + 1] = (__bf16)x.y; v[i + 2] = (__bf16)x.z; v[i + 3] = (__bf16)x.w;
}
__device__ __forceinline__ float4 mul4(float4 a, float4 b) {
  float4 r; r.x = a.x * b.x; r.y = a.y * b.y; r.z = a.z * b.z; r.w = a.w * b.w; return r;
}

// ---------------------------------------------------------------------------
// Guard-free bf16-WMMA GEMM (K % 32 == 0, M % 16 == 0, N % 16 == 0):
//   C[M,N] = A[M,K] (optionally col-scaled) @ W[N,K]^T + bias
// One wave = one 16x16 tile; all fragment loads are b128.
// ---------------------------------------------------------------------------
template <int K, bool HAS_SCALE>
__global__ __launch_bounds__(128)
void k_gemm(const float* __restrict__ A, const float* __restrict__ scale,
            const float* __restrict__ W, const float* __restrict__ bias,
            float* __restrict__ C, int M, int N) {
  const int wid = (blockIdx.x << 2) + (threadIdx.x >> 5);
  const int Nt = N >> 4;
  const int mt = wid / Nt, nt = wid % Nt;
  if (mt * 16 >= M) return;
  const int lane = threadIdx.x & 31;
  const int l15 = lane & 15, hi = lane >> 4;
  const float* arow = A + (size_t)(mt * 16 + l15) * K;
  const float* wrow = W + (size_t)(nt * 16 + l15) * K;
  v8f acc = {0.f,0.f,0.f,0.f,0.f,0.f,0.f,0.f};
#pragma unroll
  for (int kc = 0; kc < (K >> 5); ++kc) {
    const int k0 = kc << 5;
    const float* pa = arow + k0 + (hi << 3);
    float4 a0 = ld4(pa), a1 = ld4(pa + 4), a2 = ld4(pa + 16), a3 = ld4(pa + 20);
    if (HAS_SCALE) {
      const float* ps = scale + k0 + (hi << 3);
      a0 = mul4(a0, ld4(ps)); a1 = mul4(a1, ld4(ps + 4));
      a2 = mul4(a2, ld4(ps + 16)); a3 = mul4(a3, ld4(ps + 20));
    }
    v16bf a; put4(a, 0, a0); put4(a, 4, a1); put4(a, 8, a2); put4(a, 12, a3);
    const float* pb = wrow + k0 + (hi << 3);
    float4 b0 = ld4(pb), b1 = ld4(pb + 4), b2 = ld4(pb + 16), b3 = ld4(pb + 20);
    v16bf b; put4(b, 0, b0); put4(b, 4, b1); put4(b, 8, b2); put4(b, 12, b3);
    acc = __builtin_amdgcn_wmma_f32_16x16x32_bf16(false, a, false, b, (short)0, acc, false, false);
  }
  const int n = nt * 16 + l15;
  const float bs = bias ? bias[n] : 0.f;
  const int m0 = mt * 16 + (hi << 3);
#pragma unroll
  for (int r = 0; r < 8; ++r)
    C[(size_t)(m0 + r) * N + n] = acc[r] + bs;
}

// ---------------------------------------------------------------------------
// Prep: enc_in[(t,b),64] = [Y | U | 0-pad]   (zero-padded K=33 -> 64)
// ---------------------------------------------------------------------------
__global__ __launch_bounds__(256)
void k_prep_enc(const float* __restrict__ Y, const float* __restrict__ U,
                float* __restrict__ encpad) {
  const int m = blockIdx.x * 256 + threadIdx.x;
  if (m >= TB) return;
  const int t = m >> 4, b = m & 15;
  alignas(16) float v[64];
  v[0] = Y[b * TT + t];
  const float* u = U + ((size_t)b * TT + t) * NN;
#pragma unroll
  for (int i = 0; i < 32; i += 4) {
    float4 x = ld4(u + i);
    v[1 + i] = x.x; v[2 + i] = x.y; v[3 + i] = x.z; v[4 + i] = x.w;
  }
#pragma unroll
  for (int i = 33; i < 64; ++i) v[i] = 0.f;
  float* row = encpad + (size_t)m * 64;
#pragma unroll
  for (int i = 0; i < 16; ++i) st4(row + 4 * i, ((float4*)v)[i]);
}

// Prep: pad Wih (384x33) -> (384x64) for both directions.
__global__ __launch_bounds__(256)
void k_prep_w(const float* __restrict__ Wih_f, const float* __restrict__ Wih_b,
              float* __restrict__ Wpad_f, float* __restrict__ Wpad_b) {
  const int r = blockIdx.x * 256 + threadIdx.x;
  if (r >= 768) return;
  const float* src = (r < 384) ? Wih_f : Wih_b;
  float* dst = (r < 384) ? Wpad_f : Wpad_b;
  const int rr = (r < 384) ? r : r - 384;
  alignas(16) float v[64];
#pragma unroll
  for (int i = 0; i < 33; ++i) v[i] = src[rr * 33 + i];
#pragma unroll
  for (int i = 33; i < 64; ++i) v[i] = 0.f;
  float* row = dst + (size_t)rr * 64;
#pragma unroll
  for (int i = 0; i < 16; ++i) st4(row + 4 * i, ((float4*)v)[i]);
}

// ---------------------------------------------------------------------------
// Persistent GRU recurrence. grid=2 (one workgroup per direction), 256 thr =
// 8 waves. Whh lives in registers as 12 WMMA B-fragments per wave; h in LDS
// (fp32 + bf16 mirror, b128 LDS loads). 12 WMMAs/wave/step.
// ---------------------------------------------------------------------------
__global__ __launch_bounds__(256)
void k_gru(const float* __restrict__ gi_f, const float* __restrict__ gi_b,
           const float* __restrict__ Whh_f, const float* __restrict__ Whh_b,
           const float* __restrict__ bhh_f, const float* __restrict__ bhh_b,
           float* __restrict__ hf, float* __restrict__ hb) {
  const int dir = blockIdx.x;
  const float* gi  = dir ? gi_b  : gi_f;
  const float* Whh = dir ? Whh_b : Whh_f;
  const float* bhh = dir ? bhh_b : bhh_f;
  float* hout = dir ? hb : hf;

  __shared__ __bf16 h_bf[16][128];
  __shared__ float  h_f32[16][128];
  __shared__ float  gh[16][384];

  const int tid = threadIdx.x;
  for (int i = tid; i < 2048; i += 256) {
    h_f32[i >> 7][i & 127] = 0.f;
    h_bf [i >> 7][i & 127] = (__bf16)0.f;
  }

  const int lane = tid & 31;
  const int wave = tid >> 5;          // 0..7, owns N-tiles 3w..3w+2
  const int l15 = lane & 15, hi = lane >> 4;

  v16bf bfrag[3][4];
#pragma unroll
  for (int ti = 0; ti < 3; ++ti) {
    const float* wrow = Whh + (size_t)((wave * 3 + ti) * 16 + l15) * RR;
#pragma unroll
    for (int kc = 0; kc < 4; ++kc) {
      const float* p = wrow + (kc << 5) + (hi << 3);
      v16bf b;
      put4(b, 0, ld4(p)); put4(b, 4, ld4(p + 4));
      put4(b, 8, ld4(p + 16)); put4(b, 12, ld4(p + 20));
      bfrag[ti][kc] = b;
    }
  }
  __syncthreads();

  const int gb = tid >> 4;            // gate phase: batch row
  const int gn = (tid & 15) << 3;     // gate phase: hidden base (8/thread)
  const v8bf* hrow = (const v8bf*)(&h_bf[l15][0]);

  for (int step = 0; step < TT; ++step) {
    const int t = dir ? (TT - 1 - step) : step;
    v8f acc0 = {0.f,0.f,0.f,0.f,0.f,0.f,0.f,0.f};
    v8f acc1 = acc0, acc2 = acc0;
#pragma unroll
    for (int kc = 0; kc < 4; ++kc) {
      const int g = (kc << 2) + hi;
      v8bf lo = hrow[g], hh = hrow[g + 2];
      v16bf a;
#pragma unroll
      for (int j = 0; j < 8; ++j) { a[j] = lo[j]; a[j + 8] = hh[j]; }
      acc0 = __builtin_amdgcn_wmma_f32_16x16x32_bf16(false, a, false, bfrag[0][kc], (short)0, acc0, false, false);
      acc1 = __builtin_amdgcn_wmma_f32_16x16x32_bf16(false, a, false, bfrag[1][kc], (short)0, acc1, false, false);
      acc2 = __builtin_amdgcn_wmma_f32_16x16x32_bf16(false, a, false, bfrag[2][kc], (short)0, acc2, false, false);
    }
#pragma unroll
    for (int ti = 0; ti < 3; ++ti) {
      const v8f acc = (ti == 0) ? acc0 : ((ti == 1) ? acc1 : acc2);
      const int n = (wave * 3 + ti) * 16 + l15;
      const float bv = bhh[n];
      const int mb = hi << 3;
#pragma unroll
      for (int r = 0; r < 8; ++r) gh[mb + r][n] = acc[r] + bv;
    }
    __syncthreads();
    const float* gir = gi + ((size_t)t * BB + gb) * G3;
    float* ho = hout + ((size_t)t * BB + gb) * RR;
    alignas(16) float gig[24];
    ((float4*)gig)[0] = ld4(gir + gn);        ((float4*)gig)[1] = ld4(gir + gn + 4);
    ((float4*)gig)[2] = ld4(gir + 128 + gn);  ((float4*)gig)[3] = ld4(gir + 132 + gn);
    ((float4*)gig)[4] = ld4(gir + 256 + gn);  ((float4*)gig)[5] = ld4(gir + 260 + gn);
    alignas(16) float hvv[8];
    v8bf hb8;
#pragma unroll
    for (int j = 0; j < 8; ++j) {
      const int n = gn + j;
      float r = sigmoidf_(gig[j]      + gh[gb][n]);
      float z = sigmoidf_(gig[8 + j]  + gh[gb][128 + n]);
      float c = tanhf(gig[16 + j] + r * gh[gb][256 + n]);
      float hp = h_f32[gb][n];
      float hv = (1.f - z) * c + z * hp;
      hvv[j] = hv; hb8[j] = (__bf16)hv;
    }
    st4(&h_f32[gb][gn], ((float4*)hvv)[0]);
    st4(&h_f32[gb][gn + 4], ((float4*)hvv)[1]);
    *(v8bf*)(&h_bf[gb][gn]) = hb8;
    st4(ho + gn, ((float4*)hvv)[0]);
    st4(ho + gn + 4, ((float4*)hvv)[1]);
    __syncthreads();
  }
}

// ---------------------------------------------------------------------------
// Encoder projection: X[(b,t),256] = [hf|hb]@enc_W^T + enc_b.
// K-chunks 0..3 read hf, 4..7 read hb (uniform per chunk, compile-time).
// ---------------------------------------------------------------------------
__global__ __launch_bounds__(128)
void k_gemm_enc(const float* __restrict__ hf, const float* __restrict__ hb,
                const float* __restrict__ enc_W, const float* __restrict__ enc_b,
                float* __restrict__ X) {
  const int wid = (blockIdx.x << 2) + (threadIdx.x >> 5);
  const int mt = wid >> 4, nt = wid & 15;
  if (mt >= 1000) return;
  const int lane = threadIdx.x & 31;
  const int l15 = lane & 15, hi = lane >> 4;
  const int mi = mt * 16 + l15;
  const int bb = mi / TT, tt = mi % TT;
  const float* hfr = hf + ((size_t)tt * BB + bb) * RR;
  const float* hbr = hb + ((size_t)tt * BB + bb) * RR;
  const float* wrow = enc_W + (size_t)(nt * 16 + l15) * 256;
  v8f acc = {0.f,0.f,0.f,0.f,0.f,0.f,0.f,0.f};
#pragma unroll
  for (int kc = 0; kc < 8; ++kc) {
    const int k0 = kc << 5;
    const float* hsrc = (kc < 4) ? hfr : hbr;
    const int kb = k0 - ((kc < 4) ? 0 : 128);
    const float* pa = hsrc + kb + (hi << 3);
    v16bf a;
    put4(a, 0, ld4(pa)); put4(a, 4, ld4(pa + 4));
    put4(a, 8, ld4(pa + 16)); put4(a, 12, ld4(pa + 20));
    const float* pb = wrow + k0 + (hi << 3);
    v16bf b;
    put4(b, 0, ld4(pb)); put4(b, 4, ld4(pb + 4));
    put4(b, 8, ld4(pb + 16)); put4(b, 12, ld4(pb + 20));
    acc = __builtin_amdgcn_wmma_f32_16x16x32_bf16(false, a, false, b, (short)0, acc, false, false);
  }
  const int n = nt * 16 + l15;
  const float bs = enc_b[n];
  const int m0 = mt * 16 + (hi << 3);
#pragma unroll
  for (int r = 0; r < 8; ++r)
    X[(size_t)(m0 + r) * 256 + n] = acc[r] + bs;
}

// ---------------------------------------------------------------------------
// calculate_AB: per (b,t,n) thread computes log_B[K] and (t>0) log_A[K,K]
// normalized over k (axis -2) with temperature.
// ---------------------------------------------------------------------------
__global__ __launch_bounds__(256)
void k_ab(const float* __restrict__ X, const float* __restrict__ U,
          const float* __restrict__ Wzx, const float* __restrict__ Wxx,
          const float* __restrict__ Wxu, const float* __restrict__ bx,
          const int* __restrict__ temp_p,
          float* __restrict__ logB, float* __restrict__ logA) {
  const int idx = blockIdx.x * 256 + threadIdx.x;     // bt*32 + n
  const int n = idx & 31;
  const int bt = idx >> 5;
  const int t = bt % TT;
  const float invt = 1.f / (float)temp_p[0];
  alignas(16) float Xc[8], Xo[8];
  ((float4*)Xc)[0] = ld4(X + (size_t)idx * 8);
  ((float4*)Xc)[1] = ld4(X + (size_t)idx * 8 + 4);
  if (t > 0) {
    ((float4*)Xo)[0] = ld4(X + (size_t)(idx - 32) * 8);
    ((float4*)Xo)[1] = ld4(X + (size_t)(idx - 32) * 8 + 4);
  } else {
#pragma unroll
    for (int i = 0; i < 8; ++i) Xo[i] = 0.f;
  }
  const float u = U[(size_t)bt * 32 + n];
  const float* wxx = Wxx + n * 512;
  const float* wzx = Wzx + n * 512;
  const float* wxu = Wxu + n * 64;
  const float* bxn = bx  + n * 64;
  alignas(16) float lbv[8];
#pragma unroll
  for (int k = 0; k < 8; ++k) {
    float s = 0.f;
#pragma unroll
    for (int i = 0; i < 8; ++i) {
      float e = Xo[i] + wxu[k * 8 + i] * u + bxn[k * 8 + i];
#pragma unroll
      for (int j = 0; j < 8; ++j) e += wxx[(k * 8 + i) * 8 + j] * Xo[j];
      float d = e - Xc[i];
      s += d * d;
    }
    lbv[k] = -s;
  }
  float* lbo = logB + (size_t)idx * 8;
  st4(lbo, ((float4*)lbv)[0]); st4(lbo + 4, ((float4*)lbv)[1]);
  if (t > 0) {
    alignas(16) float ar[64];
#pragma unroll
    for (int k = 0; k < 8; ++k)
#pragma unroll
      for (int j = 0; j < 8; ++j) {
        float e = wxu[k * 8 + j] * u + bxn[k * 8 + j];
#pragma unroll
        for (int h = 0; h < 8; ++h) e += wzx[(k * 8 + j) * 8 + h] * Xo[h];
        ar[k * 8 + j] = e * invt;
      }
#pragma unroll
    for (int j = 0; j < 8; ++j) {
      float mx = ar[j];
#pragma unroll
      for (int k = 1; k < 8; ++k) mx = fmaxf(mx, ar[k * 8 + j]);
      float se = 0.f;
#pragma unroll
      for (int k = 0; k < 8; ++k) se += __expf(ar[k * 8 + j] - mx);
      float l = mx + __logf(se);
#pragma unroll
      for (int k = 0; k < 8; ++k) ar[k * 8 + j] -= l;
    }
    float* lao = logA + (size_t)idx * 64;
#pragma unroll
    for (int q = 0; q < 16; ++q) st4(lao + 4 * q, ((float4*)ar)[q]);
  }
}

// ---------------------------------------------------------------------------
// Forward message passing: 512 independent (b,n) chains, state in registers.
// ---------------------------------------------------------------------------
__global__ void k_fwd(const float* __restrict__ Zinit, const float* __restrict__ logB,
                      const float* __restrict__ logA, float* __restrict__ fwd) {
  const int tid = blockIdx.x * 32 + threadIdx.x;
  if (tid >= 512) return;
  const int b = tid >> 5, n = tid & 31;
  const size_t base = (size_t)b * TT * 32 + n;
  alignas(16) float p[8];
  {
    alignas(16) float lb[8];
    ((float4*)lb)[0] = ld4(logB + base * 8);
    ((float4*)lb)[1] = ld4(logB + base * 8 + 4);
    float mx = -1e30f;
#pragma unroll
    for (int k = 0; k < 8; ++k) { p[k] = Zinit[n * 8 + k] + lb[k]; mx = fmaxf(mx, p[k]); }
    float se = 0.f;
#pragma unroll
    for (int k = 0; k < 8; ++k) se += __expf(p[k] - mx);
    float l = mx + __logf(se);
#pragma unroll
    for (int k = 0; k < 8; ++k) p[k] -= l;
    float* o = fwd + base * 8;
    st4(o, ((float4*)p)[0]); st4(o + 4, ((float4*)p)[1]);
  }
  for (int t = 1; t < TT; ++t) {
    const size_t cur = base + (size_t)t * 32;
    alignas(16) float a[64], lb[8];
#pragma unroll
    for (int q = 0; q < 16; ++q) ((float4*)a)[q] = ld4(logA + cur * 64 + 4 * q);
    ((float4*)lb)[0] = ld4(logB + cur * 8);
    ((float4*)lb)[1] = ld4(logB + cur * 8 + 4);
    alignas(16) float q8[8]; float mx2 = -1e30f;
#pragma unroll
    for (int k = 0; k < 8; ++k) {
      float m = -1e30f, v[8];
#pragma unroll
      for (int j = 0; j < 8; ++j) { v[j] = a[k * 8 + j] + p[j]; m = fmaxf(m, v[j]); }
      float s = 0.f;
#pragma unroll
      for (int j = 0; j < 8; ++j) s += __expf(v[j] - m);
      q8[k] = lb[k] + m + __logf(s);
      mx2 = fmaxf(mx2, q8[k]);
    }
    float se = 0.f;
#pragma unroll
    for (int k = 0; k < 8; ++k) se += __expf(q8[k] - mx2);
    float l = mx2 + __logf(se);
#pragma unroll
    for (int k = 0; k < 8; ++k) { q8[k] -= l; p[k] = q8[k]; }
    float* o = fwd + cur * 8;
    st4(o, ((float4*)q8)[0]); st4(o + 4, ((float4*)q8)[1]);
  }
}

// ---------------------------------------------------------------------------
// Backward message passing.
// ---------------------------------------------------------------------------
__global__ void k_bwd(const float* __restrict__ logB, const float* __restrict__ logA,
                      float* __restrict__ bwd) {
  const int tid = blockIdx.x * 32 + threadIdx.x;
  if (tid >= 512) return;
  const int b = tid >> 5, n = tid & 31;
  alignas(16) float p[8];
#pragma unroll
  for (int k = 0; k < 8; ++k) p[k] = 0.f;
  {
    float* o = bwd + (((size_t)b * TT + (TT - 1)) * 32 + n) * 8;
    st4(o, ((float4*)p)[0]); st4(o + 4, ((float4*)p)[1]);
  }
  for (int t = TT - 1; t >= 1; --t) {
    const size_t cur = ((size_t)b * TT + t) * 32 + n;
    alignas(16) float a[64], lb[8];
#pragma unroll
    for (int q = 0; q < 16; ++q) ((float4*)a)[q] = ld4(logA + cur * 64 + 4 * q);
    ((float4*)lb)[0] = ld4(logB + cur * 8);
    ((float4*)lb)[1] = ld4(logB + cur * 8 + 4);
    float pk[8];
#pragma unroll
    for (int k = 0; k < 8; ++k) pk[k] = p[k] + lb[k];
    alignas(16) float q8[8]; float mx2 = -1e30f;
#pragma unroll
    for (int j = 0; j < 8; ++j) {
      float m = -1e30f, v[8];
#pragma unroll
      for (int k = 0; k < 8; ++k) { v[k] = pk[k] + a[k * 8 + j]; m = fmaxf(m, v[k]); }
      float s = 0.f;
#pragma unroll
      for (int k = 0; k < 8; ++k) s += __expf(v[k] - m);
      q8[j] = m + __logf(s);
      mx2 = fmaxf(mx2, q8[j]);
    }
    float se = 0.f;
#pragma unroll
    for (int j = 0; j < 8; ++j) se += __expf(q8[j] - mx2);
    float l = mx2 + __logf(se);
#pragma unroll
    for (int j = 0; j < 8; ++j) { q8[j] -= l; p[j] = q8[j]; }
    float* o = bwd + (cur - 32) * 8;
    st4(o, ((float4*)q8)[0]); st4(o + 4, ((float4*)q8)[1]);
  }
}

__global__ void k_zero(float* __restrict__ out) {
  if (threadIdx.x < 4) out[threadIdx.x] = 0.f;
}

// ---------------------------------------------------------------------------
// Final reduction: g1/g2 posteriors -> init_prob, seq_prob, gamma_prior.
// ---------------------------------------------------------------------------
__global__ __launch_bounds__(256)
void k_final(const float* __restrict__ fwd, const float* __restrict__ bwd,
             const float* __restrict__ logA, const float* __restrict__ logB,
             const float* __restrict__ Zinit, float* __restrict__ out) {
  __shared__ float sred[3];
  if (threadIdx.x == 0) { sred[0] = 0.f; sred[1] = 0.f; sred[2] = 0.f; }
  __syncthreads();
  const int idx = blockIdx.x * 256 + threadIdx.x;   // bt*32 + n
  const int n = idx & 31;
  const int bt = idx >> 5;
  const int t = bt % TT;
  alignas(16) float fp[8], wp[8], lb[8];
  ((float4*)fp)[0] = ld4(fwd + (size_t)idx * 8);  ((float4*)fp)[1] = ld4(fwd + (size_t)idx * 8 + 4);
  ((float4*)wp)[0] = ld4(bwd + (size_t)idx * 8);  ((float4*)wp)[1] = ld4(bwd + (size_t)idx * 8 + 4);
  ((float4*)lb)[0] = ld4(logB + (size_t)idx * 8); ((float4*)lb)[1] = ld4(logB + (size_t)idx * 8 + 4);
  float s[8]; float mx = -1e30f;
#pragma unroll
  for (int k = 0; k < 8; ++k) { s[k] = fp[k] + wp[k]; mx = fmaxf(mx, s[k]); }
  float se = 0.f;
#pragma unroll
  for (int k = 0; k < 8; ++k) se += __expf(s[k] - mx);
  const float l1 = mx + __logf(se);
  float gp = 0.f, ini = 0.f, seq = 0.f;
#pragma unroll
  for (int k = 0; k < 8; ++k) gp += (s[k] - l1);
  gp *= 0.125f;                                       // /K prior
  if (t == 0) {
#pragma unroll
    for (int k = 0; k < 8; ++k)
      ini += __expf(s[k] - l1) * (lb[k] + Zinit[n * 8 + k]);
  } else {
    alignas(16) float fprev[8], a[64];
    ((float4*)fprev)[0] = ld4(fwd + (size_t)(idx - 32) * 8);
    ((float4*)fprev)[1] = ld4(fwd + (size_t)(idx - 32) * 8 + 4);
#pragma unroll
    for (int q = 0; q < 16; ++q) ((float4*)a)[q] = ld4(logA + (size_t)idx * 64 + 4 * q);
    float m[64]; float m2 = -1e30f;
#pragma unroll
    for (int k = 0; k < 8; ++k)
#pragma unroll
      for (int j = 0; j < 8; ++j) {
        float v = fprev[j] + wp[k] + a[k * 8 + j] + lb[k];
        m[k * 8 + j] = v;
        m2 = fmaxf(m2, v);
      }
    float ss = 0.f;
#pragma unroll
    for (int i = 0; i < 64; ++i) ss += __expf(m[i] - m2);
    const float L = m2 + __logf(ss);
#pragma unroll
    for (int k = 0; k < 8; ++k)
#pragma unroll
      for (int j = 0; j < 8; ++j)
        seq += __expf(m[k * 8 + j] - L) * (a[k * 8 + j] + lb[k]);
  }
  atomicAdd(&sred[0], ini);
  atomicAdd(&sred[1], seq);
  atomicAdd(&sred[2], gp);
  __syncthreads();
  if (threadIdx.x == 0) {
    atomicAdd(&out[1], sred[0]);
    atomicAdd(&out[2], sred[1]);
    atomicAdd(&out[3], sred[2]);
  }
}

__global__ __launch_bounds__(256)
void k_yprob(const float* __restrict__ X, const float* __restrict__ Wy,
             const float* __restrict__ Wyb, const float* __restrict__ Y,
             float* __restrict__ out) {
  __shared__ float sred;
  if (threadIdx.x == 0) sred = 0.f;
  __syncthreads();
  const int bt = blockIdx.x * 256 + threadIdx.x;
  float loc = 0.f;
  if (bt < TB) {
    const float* x = X + (size_t)bt * 256;
    float acc = Wyb[0];
#pragma unroll 4
    for (int i = 0; i < 256; i += 4) {
      float4 xv = ld4(x + i), wv = ld4(Wy + i);
      acc += xv.x * wv.x + xv.y * wv.y + xv.z * wv.z + xv.w * wv.w;
    }
    const float d = acc - Y[bt];
    loc = -d * d;
  }
  atomicAdd(&sred, loc);
  __syncthreads();
  if (threadIdx.x == 0) atomicAdd(&out[0], sred);
}

// ---------------------------------------------------------------------------
extern "C" void kernel_launch(void* const* d_in, const int* in_sizes, int n_in,
                              void* d_out, int out_size, void* d_ws, size_t ws_size,
                              hipStream_t stream) {
  const float* Y       = (const float*)d_in[0];
  const float* U_raw   = (const float*)d_in[1];
  const float* C_syn   = (const float*)d_in[2];
  const float* U_scale = (const float*)d_in[3];
  const float* Z_init  = (const float*)d_in[4];
  const float* W_zx    = (const float*)d_in[5];
  const float* W_xx    = (const float*)d_in[6];
  const float* W_xu    = (const float*)d_in[7];
  const float* b_x     = (const float*)d_in[8];
  const float* W_yx_w  = (const float*)d_in[9];
  const float* W_yx_b  = (const float*)d_in[10];
  const float* Wih_f   = (const float*)d_in[11];
  const float* Whh_f   = (const float*)d_in[12];
  const float* bih_f   = (const float*)d_in[13];
  const float* bhh_f   = (const float*)d_in[14];
  const float* Wih_b   = (const float*)d_in[15];
  const float* Whh_b   = (const float*)d_in[16];
  const float* bih_b   = (const float*)d_in[17];
  const float* bhh_b   = (const float*)d_in[18];
  const float* enc_W   = (const float*)d_in[19];
  const float* enc_b   = (const float*)d_in[20];
  const int*   temp_p  = (const int*)d_in[21];
  (void)in_sizes; (void)n_in; (void)out_size; (void)ws_size;

  char* ws = (char*)d_ws;
  size_t off = 0;
  auto take = [&](size_t bytes) -> char* {
    char* p = ws + off;
    off += (bytes + 255) & ~(size_t)255;
    return p;
  };
  float* U      = (float*)take((size_t)TB * NN * 4);
  float* encpad = (float*)take((size_t)TB * 64 * 4);
  float* Wpad_f = (float*)take((size_t)G3 * 64 * 4);
  float* Wpad_b = (float*)take((size_t)G3 * 64 * 4);
  float* gi_f   = (float*)take((size_t)TB * G3 * 4);
  float* gi_b   = (float*)take((size_t)TB * G3 * 4);
  float* hf     = (float*)take((size_t)TB * RR * 4);
  float* hb     = (float*)take((size_t)TB * RR * 4);
  float* X      = (float*)take((size_t)TB * NN * HH * 4);
  float* logB   = (float*)take((size_t)TB * NN * KK * 4);
  float* logA   = (float*)take((size_t)TB * NN * KK * KK * 4);
  float* fwd    = (float*)take((size_t)TB * NN * KK * 4);
  float* bwd    = (float*)take((size_t)TB * NN * KK * 4);
  float* out    = (float*)d_out;

  // 1) U = (U_raw * U_scale) @ C_syn^T           (guard-free WMMA, K=256)
  k_gemm<256, true><<<500, 128, 0, stream>>>(U_raw, U_scale, C_syn, nullptr, U, TB, NN);
  // 2) pad [Y|U] -> enc_in (K=64) and Wih -> 384x64
  k_prep_enc<<<63, 256, 0, stream>>>(Y, U, encpad);
  k_prep_w<<<3, 256, 0, stream>>>(Wih_f, Wih_b, Wpad_f, Wpad_b);
  // 3) GRU input projections (guard-free WMMA, K=64)
  k_gemm<64, false><<<6000, 128, 0, stream>>>(encpad, nullptr, Wpad_f, bih_f, gi_f, TB, G3);
  k_gemm<64, false><<<6000, 128, 0, stream>>>(encpad, nullptr, Wpad_b, bih_b, gi_b, TB, G3);
  // 4) Persistent bidirectional GRU recurrence (register-resident Whh WMMA)
  k_gru<<<2, 256, 0, stream>>>(gi_f, gi_b, Whh_f, Whh_b, bhh_f, bhh_b, hf, hb);
  // 5) Encoder projection X = [hf|hb]@enc_W^T + enc_b (WMMA)
  k_gemm_enc<<<4000, 128, 0, stream>>>(hf, hb, enc_W, enc_b, X);
  // 6) log_A / log_B
  k_ab<<<2000, 256, 0, stream>>>(X, U, W_zx, W_xx, W_xu, b_x, temp_p, logB, logA);
  // 7) forward / backward message passing
  k_fwd<<<16, 32, 0, stream>>>(Z_init, logB, logA, fwd);
  k_bwd<<<16, 32, 0, stream>>>(logB, logA, bwd);
  // 8) scalar outputs
  k_zero<<<1, 32, 0, stream>>>(out);
  k_final<<<2000, 256, 0, stream>>>(fwd, bwd, logA, logB, Z_init, out);
  k_yprob<<<63, 256, 0, stream>>>(X, W_yx_w, W_yx_b, Y, out);
}
